// SimpleGlue_29944511988267
// MI455X (gfx1250) — matrix operand order repaired and checked by reading, not compiled, not run on hardware
//
#include <hip/hip_runtime.h>
#include <hip/hip_bf16.h>

// ---------------------------------------------------------------------------
// Problem constants (fixed by the reference)
// ---------------------------------------------------------------------------
#define BATCH   2
#define SEQ     1024
#define DIM     256
#define HEADS   4
#define HDIM    64
#define LAYERS  9
#define ROWS    (BATCH * SEQ)       // 2048

typedef __attribute__((ext_vector_type(16))) _Float16     v16h;
typedef __attribute__((ext_vector_type(8)))  float        v8f;
typedef __attribute__((ext_vector_type(4)))  unsigned int v4u;
typedef __attribute__((ext_vector_type(4)))  int          v4i;
typedef __attribute__((ext_vector_type(8)))  int          v8i;

#define HAVE_TDM (__has_builtin(__builtin_amdgcn_tensor_load_to_lds) && \
                  __has_builtin(__builtin_amdgcn_s_wait_tensorcnt))

__device__ __forceinline__ v8f wmma16x16x32(v16h a, v16h b, v8f c) {
    return __builtin_amdgcn_wmma_f32_16x16x32_f16(
        /*neg_a=*/false, a, /*neg_b=*/false, b,
        /*c_mod=*/(short)0, c, /*reuse_a=*/false, /*reuse_b=*/false);
}

// A-fragment from a contiguous f16 row (lane half `hi` holds
// K = kc + hi*16 + j  for j = 0..15).
__device__ __forceinline__ v16h afrag_f16(const _Float16* __restrict__ row, int kc, int hi) {
    v16h a;
    const _Float16* p0 = row + kc + hi * 16;
    const _Float16* p1 = row + kc + 8 + hi * 16;
#pragma unroll
    for (int j = 0; j < 8; ++j) { a[j] = p0[j]; a[8 + j] = p1[j]; }
    return a;
}

// A-fragment built on the fly from an f32 row (convert to f16).
__device__ __forceinline__ v16h afrag_f32(const float* __restrict__ row, int kc, int hi) {
    v16h a;
    const float4* p0 = (const float4*)(row + kc + hi * 16);
    const float4* p1 = (const float4*)(row + kc + 8 + hi * 16);
    float4 f;
    f = p0[0]; a[0]  = (_Float16)f.x; a[1]  = (_Float16)f.y; a[2]  = (_Float16)f.z; a[3]  = (_Float16)f.w;
    f = p0[1]; a[4]  = (_Float16)f.x; a[5]  = (_Float16)f.y; a[6]  = (_Float16)f.z; a[7]  = (_Float16)f.w;
    f = p1[0]; a[8]  = (_Float16)f.x; a[9]  = (_Float16)f.y; a[10] = (_Float16)f.z; a[11] = (_Float16)f.w;
    f = p1[1]; a[12] = (_Float16)f.x; a[13] = (_Float16)f.y; a[14] = (_Float16)f.z; a[15] = (_Float16)f.w;
    return a;
}

// ---------------------------------------------------------------------------
// WMMA GEMM:  C[M x N] = (A[M x K](f32) @ W (+ bias)) * scale (+ resid)
// W is pre-transposed f16: WT[n*K + k].  Block = 8 waves = 128x64 output;
// each wave owns a 16x64 strip (4 accumulators sharing one A fragment).
// The 64 x K f16 weight panel (contiguous in WT) is staged into LDS via the
// Tensor Data Mover when available, else a cooperative vectorized copy.
// ---------------------------------------------------------------------------
__global__ __launch_bounds__(256)
void k_gemm(const float* __restrict__ A, const _Float16* __restrict__ WT,
            const float* __restrict__ bias, float* __restrict__ C,
            const float* __restrict__ resid,
            int M, int N, int K, int ldc, float scale)
{
    __shared__ __align__(128) _Float16 Bs[64 * 512];   // 64 cols x K (K <= 512)
    const int lane = threadIdx.x & 31;
    const int wave = threadIdx.x >> 5;
    const int lo = lane & 15, hi = lane >> 4;
    const int tn_blk = blockIdx.x * 64;
    const int tm = blockIdx.y * 128 + wave * 16;

    const _Float16* wsrc = WT + (size_t)tn_blk * K;    // 64*K contiguous halves
#if HAVE_TDM
    if (wave == 0) {
        unsigned long long ga = (unsigned long long)(uintptr_t)wsrc;
        unsigned int lds = (unsigned int)(uintptr_t)&Bs[0];
        // D# group 0: count=1 | lds_addr | global_addr | type=2 ("image")
        v4u g0;
        g0[0] = 1u;
        g0[1] = lds;
        g0[2] = (unsigned int)(ga & 0xFFFFFFFFu);
        g0[3] = (unsigned int)((ga >> 32) & 0x01FFFFFFu) | (2u << 30);
        // D# group 1: data_size=2B; tensor 2D: dim0=K (row), dim1=64 rows;
        // tile K x 64; dim0_stride = K, dim1_stride = K.
        v8i g1;
        g1[0] = 0x00010000;                    // wg_mask=0, data_size=1 (2 bytes)
        g1[1] = (K & 0xFFFF) << 16;            // tensor_dim0[15:0] in bits 63:48
        g1[2] = (64 << 16);                    // tensor_dim0 hi=0 | tensor_dim1 lo=64
        g1[3] = (K & 0xFFFF) << 16;            // tensor_dim1 hi=0 | tile_dim0=K
        g1[4] = 64;                            // tile_dim1=64, tile_dim2=0
        g1[5] = K;                             // tensor_dim0_stride lo
        g1[6] = (K & 0xFFFF) << 16;            // stride0 hi=0 | tensor_dim1_stride lo
        g1[7] = 0;
        v4i z4 = {0, 0, 0, 0};
#if __clang_major__ >= 23
        v8i z8 = {0, 0, 0, 0, 0, 0, 0, 0};
        __builtin_amdgcn_tensor_load_to_lds(g0, g1, z4, z4, z8, 0);
#else
        __builtin_amdgcn_tensor_load_to_lds(g0, g1, z4, z4, 0);
#endif
        __builtin_amdgcn_s_wait_tensorcnt(0);
    }
#else
    {
        const uint4* src = (const uint4*)wsrc;
        uint4* dst = (uint4*)Bs;
        const int total = (64 * K) >> 3;       // uint4 = 8 halves
        for (int i = threadIdx.x; i < total; i += 256) dst[i] = src[i];
    }
#endif
    __syncthreads();

    v8f acc[4];
#pragma unroll
    for (int c = 0; c < 4; ++c) {
        float bv = bias ? bias[tn_blk + c * 16 + lo] : 0.0f;
#pragma unroll
        for (int i = 0; i < 8; ++i) acc[c][i] = bv;
    }

    const float* arow = A + (size_t)(tm + lo) * K;
    for (int kc = 0; kc < K; kc += 32) {
        __builtin_prefetch(arow + kc + 64, 0, 0);
        v16h a = afrag_f32(arow, kc, hi);
#pragma unroll
        for (int c = 0; c < 4; ++c) {
            const _Float16* brow = Bs + (size_t)(c * 16 + lo) * K;
            v16h b = *(const v16h*)(brow + kc + hi * 16);
            acc[c] = wmma16x16x32(a, b, acc[c]);
        }
    }

#pragma unroll
    for (int c = 0; c < 4; ++c) {
        float* cptr = C + (size_t)(tm + hi * 8) * ldc + (tn_blk + c * 16 + lo);
        const float* rptr = resid ? resid + (size_t)(tm + hi * 8) * ldc + (tn_blk + c * 16 + lo)
                                  : nullptr;
#pragma unroll
        for (int v = 0; v < 8; ++v) {
            float val = acc[c][v] * scale;
            if (resid) val += rptr[(size_t)v * ldc];
            cptr[(size_t)v * ldc] = val;
        }
    }
}

// ---------------------------------------------------------------------------
// Flash attention: Out[b, s, h*64 + d] = softmax(Q K^T * scale) V
// Qh, Kh: f16 [B*H][S][64];  Vt: f16 [B*H][64][S] (pre-transposed).
// One wave handles one (b, h, 16-row Q tile); online softmax in LDS.
// ---------------------------------------------------------------------------
__global__ __launch_bounds__(256)
void k_attn(const _Float16* __restrict__ Qh, const _Float16* __restrict__ Kh,
            const _Float16* __restrict__ Vt, float* __restrict__ Out, float scale)
{
    const int lane = threadIdx.x & 31;
    const int wave = threadIdx.x >> 5;
    const int qtiles = SEQ / 16;                 // 64
    const int task = blockIdx.x * 8 + wave;      // 0 .. B*H*qtiles-1
    if (task >= BATCH * HEADS * qtiles) return;
    const int bh = task / qtiles;
    const int qt = task % qtiles;
    const int b = bh >> 2, h = bh & 3;
    const int lo = lane & 15, hi = lane >> 4;

    __shared__ __align__(32) float    Sl[8][16][33];
    __shared__ __align__(32) _Float16 Pl[8][16][48];
    __shared__ float rmax[8][16], rsum[8][16], ralpha[8][16];

    const _Float16* qrow = Qh + ((size_t)bh * SEQ + qt * 16 + lo) * HDIM;
    v16h aq0 = afrag_f16(qrow, 0, hi);
    v16h aq1 = afrag_f16(qrow, 32, hi);

    v8f o0 = {}, o1 = {}, o2 = {}, o3 = {};
    if (lane < 16) { rmax[wave][lane] = -3.0e30f; rsum[wave][lane] = 0.0f; }
    __builtin_amdgcn_wave_barrier();

    for (int kt = 0; kt < SEQ; kt += 32) {
        const _Float16* kbase = Kh + ((size_t)bh * SEQ + kt) * HDIM;
        v8f s0 = {}, s1 = {};
        {
            v16h b0 = *(const v16h*)(kbase + (size_t)lo * HDIM + hi * 16);
            v16h b1 = *(const v16h*)(kbase + (size_t)(16 + lo) * HDIM + hi * 16);
            s0 = wmma16x16x32(aq0, b0, s0);
            s1 = wmma16x16x32(aq0, b1, s1);
            b0 = *(const v16h*)(kbase + (size_t)lo * HDIM + 32 + hi * 16);
            b1 = *(const v16h*)(kbase + (size_t)(16 + lo) * HDIM + 32 + hi * 16);
            s0 = wmma16x16x32(aq1, b0, s0);
            s1 = wmma16x16x32(aq1, b1, s1);
        }
#pragma unroll
        for (int v = 0; v < 8; ++v) {
            Sl[wave][v + 8 * hi][lo]      = s0[v] * scale;
            Sl[wave][v + 8 * hi][16 + lo] = s1[v] * scale;
        }
        __builtin_amdgcn_wave_barrier();

        if (lane < 16) {
            float mo = rmax[wave][lane];
            float tm = mo;
#pragma unroll 4
            for (int j = 0; j < 32; ++j) tm = fmaxf(tm, Sl[wave][lane][j]);
            float al  = __expf(mo - tm);
            float sum = rsum[wave][lane] * al;
            for (int j = 0; j < 32; ++j) {
                float p = __expf(Sl[wave][lane][j] - tm);
                Pl[wave][lane][j] = (_Float16)p;
                sum += p;
            }
            rmax[wave][lane] = tm; rsum[wave][lane] = sum; ralpha[wave][lane] = al;
        }
        __builtin_amdgcn_wave_barrier();

#pragma unroll
        for (int v = 0; v < 8; ++v) {
            float al = ralpha[wave][v + 8 * hi];
            o0[v] *= al; o1[v] *= al; o2[v] *= al; o3[v] *= al;
        }
        v16h ap = afrag_f16(&Pl[wave][lo][0], 0, hi);
        const _Float16* vb = Vt + (size_t)bh * HDIM * SEQ + kt;
        o0 = wmma16x16x32(ap, *(const v16h*)(vb + (size_t)(0 * 16 + lo) * SEQ + hi * 16), o0);
        o1 = wmma16x16x32(ap, *(const v16h*)(vb + (size_t)(1 * 16 + lo) * SEQ + hi * 16), o1);
        o2 = wmma16x16x32(ap, *(const v16h*)(vb + (size_t)(2 * 16 + lo) * SEQ + hi * 16), o2);
        o3 = wmma16x16x32(ap, *(const v16h*)(vb + (size_t)(3 * 16 + lo) * SEQ + hi * 16), o3);
    }

    float* ob = Out + ((size_t)b * SEQ + qt * 16) * DIM + h * HDIM;
#pragma unroll
    for (int v = 0; v < 8; ++v) {
        int m = v + 8 * hi;
        float inv = 1.0f / rsum[wave][m];
        ob[(size_t)m * DIM + 0  + lo] = o0[v] * inv;
        ob[(size_t)m * DIM + 16 + lo] = o1[v] * inv;
        ob[(size_t)m * DIM + 32 + lo] = o2[v] * inv;
        ob[(size_t)m * DIM + 48 + lo] = o3[v] * inv;
    }
}

// ---------------------------------------------------------------------------
// Small support kernels
// ---------------------------------------------------------------------------

// f32 [K x N] -> f16 transposed [N x K]
__global__ void k_convT(const float* __restrict__ W, _Float16* __restrict__ WT, int K, int N) {
    int gid = blockIdx.x * blockDim.x + threadIdx.x;
    if (gid >= K * N) return;
    int n = gid / K, k = gid - n * K;
    WT[gid] = (_Float16)W[(size_t)k * N + n];
}

__global__ void k_conv16(const float* __restrict__ X, _Float16* __restrict__ Y, int count) {
    int gid = blockIdx.x * blockDim.x + threadIdx.x;
    if (gid < count) Y[gid] = (_Float16)X[gid];
}

// kpts normalization stats per batch: shift0, shift1, scale
__global__ void k_stats(const float* __restrict__ kpts, float* __restrict__ stats) {
    int b = blockIdx.x, tid = threadIdx.x;
    __shared__ float smn0[256], smn1[256], smx0[256], smx1[256];
    float mn0 = 3e30f, mn1 = 3e30f, mx0 = -3e30f, mx1 = -3e30f;
    for (int s = tid; s < SEQ; s += 256) {
        float x = kpts[((size_t)b * SEQ + s) * 2 + 0];
        float y = kpts[((size_t)b * SEQ + s) * 2 + 1];
        mn0 = fminf(mn0, x); mx0 = fmaxf(mx0, x);
        mn1 = fminf(mn1, y); mx1 = fmaxf(mx1, y);
    }
    smn0[tid] = mn0; smn1[tid] = mn1; smx0[tid] = mx0; smx1[tid] = mx1;
    __syncthreads();
    for (int st = 128; st > 0; st >>= 1) {
        if (tid < st) {
            smn0[tid] = fminf(smn0[tid], smn0[tid + st]);
            smn1[tid] = fminf(smn1[tid], smn1[tid + st]);
            smx0[tid] = fmaxf(smx0[tid], smx0[tid + st]);
            smx1[tid] = fmaxf(smx1[tid], smx1[tid + st]);
        }
        __syncthreads();
    }
    if (tid == 0) {
        float sz0 = 1.0f + smx0[0] - smn0[0];
        float sz1 = 1.0f + smx1[0] - smn1[0];
        stats[b * 3 + 0] = sz0 * 0.5f;
        stats[b * 3 + 1] = sz1 * 0.5f;
        stats[b * 3 + 2] = fmaxf(sz0, sz1) * 0.5f;
    }
}

// rotary embedding tables: cos/sin [B][S][64] (each proj value repeated twice)
__global__ void k_posenc(const float* __restrict__ kpts, const float* __restrict__ stats,
                         const float* __restrict__ Wr,
                         float* __restrict__ cosb, float* __restrict__ sinb) {
    int bs = blockIdx.x;            // b*SEQ + s
    int b = bs >> 10;
    int t = threadIdx.x;            // 0..63
    int i = t >> 1;                 // proj index 0..31
    float scl = stats[b * 3 + 2];
    float k0 = (kpts[(size_t)bs * 2 + 0] - stats[b * 3 + 0]) / scl;
    float k1 = (kpts[(size_t)bs * 2 + 1] - stats[b * 3 + 1]) / scl;
    float proj = k0 * Wr[i] + k1 * Wr[32 + i];
    cosb[(size_t)bs * 64 + t] = __cosf(proj);
    sinb[(size_t)bs * 64 + t] = __sinf(proj);
}

// qkv [B][S][768] (packed h,d,{q,k,v}) -> rope(q),rope(k) f16 [B*H][S][64], v f16 transposed [B*H][64][S]
__global__ void k_qkv_rope_split(const float* __restrict__ qkv, const float* __restrict__ cosb,
                                 const float* __restrict__ sinb, _Float16* __restrict__ qh,
                                 _Float16* __restrict__ kh, _Float16* __restrict__ vt) {
    int gid = blockIdx.x * blockDim.x + threadIdx.x;   // B*H*S*32 pairs
    if (gid >= BATCH * HEADS * SEQ * 32) return;
    int p = gid & 31;
    int s = (gid >> 5) & (SEQ - 1);
    int h = (gid >> 15) & 3;
    int b = gid >> 17;
    size_t base = ((size_t)(b * SEQ + s)) * 768 + h * 192 + p * 6;
    float q0 = qkv[base + 0], k0 = qkv[base + 1], v0 = qkv[base + 2];
    float q1 = qkv[base + 3], k1 = qkv[base + 4], v1 = qkv[base + 5];
    size_t eb = ((size_t)(b * SEQ + s)) * 64 + p * 2;
    float c = cosb[eb], sn = sinb[eb];
    size_t hb = ((size_t)((b * HEADS + h) * SEQ) + s) * 64 + p * 2;
    qh[hb]     = (_Float16)(q0 * c - q1 * sn);
    qh[hb + 1] = (_Float16)(q1 * c + q0 * sn);
    kh[hb]     = (_Float16)(k0 * c - k1 * sn);
    kh[hb + 1] = (_Float16)(k1 * c + k0 * sn);
    size_t vtb = ((size_t)(b * HEADS + h) * 64 + p * 2) * (size_t)SEQ + s;
    vt[vtb]       = (_Float16)v0;
    vt[vtb + SEQ] = (_Float16)v1;
}

// heads split [B][S][256] f32 -> f16 [B*H][S][64] (TR=false) or [B*H][64][S] (TR=true)
template <bool TR>
__global__ void k_heads_split(const float* __restrict__ X, _Float16* __restrict__ Y) {
    int gid = blockIdx.x * blockDim.x + threadIdx.x;   // B*H*S*64
    if (gid >= BATCH * HEADS * SEQ * 64) return;
    int d = gid & 63;
    int s = (gid >> 6) & (SEQ - 1);
    int h = (gid >> 16) & 3;
    int b = gid >> 18;
    float v = X[((size_t)(b * SEQ + s)) * DIM + h * 64 + d];
    if (TR) Y[((size_t)(b * HEADS + h) * 64 + d) * SEQ + s] = (_Float16)v;
    else    Y[(((size_t)(b * HEADS + h) * SEQ) + s) * 64 + d] = (_Float16)v;
}

// copy x [2048 x 256] into first 256 cols of cat [2048 x 512]
__global__ void k_copy_half(const float* __restrict__ X, float* __restrict__ Y) {
    int gid = blockIdx.x * blockDim.x + threadIdx.x;
    if (gid >= ROWS * DIM) return;
    int c = gid & 255, r = gid >> 8;
    Y[(size_t)r * 512 + c] = X[gid];
}

// in-place LayerNorm + exact GELU on rows of 512
__global__ __launch_bounds__(256)
void k_ln_gelu(float* __restrict__ X, const float* __restrict__ g, const float* __restrict__ bt) {
    const int row = blockIdx.x, tid = threadIdx.x;
    float* xr = X + (size_t)row * 512;
    float v0 = xr[tid], v1 = xr[tid + 256];
    __shared__ float red[256];
    red[tid] = v0 + v1;
    __syncthreads();
    for (int st = 128; st > 0; st >>= 1) { if (tid < st) red[tid] += red[tid + st]; __syncthreads(); }
    float mu = red[0] * (1.0f / 512.0f);
    __syncthreads();
    float d0 = v0 - mu, d1 = v1 - mu;
    red[tid] = d0 * d0 + d1 * d1;
    __syncthreads();
    for (int st = 128; st > 0; st >>= 1) { if (tid < st) red[tid] += red[tid + st]; __syncthreads(); }
    float rstd = rsqrtf(red[0] * (1.0f / 512.0f) + 1e-5f);
    float y0 = d0 * rstd * g[tid] + bt[tid];
    float y1 = d1 * rstd * g[tid + 256] + bt[tid + 256];
    xr[tid]       = 0.5f * y0 * (1.0f + erff(y0 * 0.70710678f));
    xr[tid + 256] = 0.5f * y1 * (1.0f + erff(y1 * 0.70710678f));
}

// softmax over rows of sim [B*1024][1024]
__global__ __launch_bounds__(256)
void k_softmax_rows(const float* __restrict__ Sm, float* __restrict__ P) {
    const int row = blockIdx.x, tid = threadIdx.x;
    const float* sr = Sm + (size_t)row * SEQ;
    float* pr = P + (size_t)row * SEQ;
    float v[4], mx = -3e30f;
#pragma unroll
    for (int i = 0; i < 4; ++i) { v[i] = sr[tid + i * 256]; mx = fmaxf(mx, v[i]); }
    __shared__ float red[256];
    red[tid] = mx;
    __syncthreads();
    for (int st = 128; st > 0; st >>= 1) { if (tid < st) red[tid] = fmaxf(red[tid], red[tid + st]); __syncthreads(); }
    mx = red[0];
    __syncthreads();
    float sum = 0.0f;
#pragma unroll
    for (int i = 0; i < 4; ++i) { v[i] = __expf(v[i] - mx); sum += v[i]; }
    red[tid] = sum;
    __syncthreads();
    for (int st = 128; st > 0; st >>= 1) { if (tid < st) red[tid] += red[tid + st]; __syncthreads(); }
    float inv = 1.0f / red[0];
#pragma unroll
    for (int i = 0; i < 4; ++i) pr[tid + i * 256] = v[i] * inv;
}

// softmax over columns of sim [b][1024][1024]
__global__ __launch_bounds__(256)
void k_softmax_cols(const float* __restrict__ Sm, float* __restrict__ P) {
    const int b = blockIdx.x >> 10, col = blockIdx.x & 1023, tid = threadIdx.x;
    const float* sc = Sm + (size_t)b * SEQ * SEQ + col;
    float* pc = P + (size_t)b * SEQ * SEQ + col;
    float v[4], mx = -3e30f;
#pragma unroll
    for (int i = 0; i < 4; ++i) { v[i] = sc[(size_t)(tid + i * 256) * SEQ]; mx = fmaxf(mx, v[i]); }
    __shared__ float red[256];
    red[tid] = mx;
    __syncthreads();
    for (int st = 128; st > 0; st >>= 1) { if (tid < st) red[tid] = fmaxf(red[tid], red[tid + st]); __syncthreads(); }
    mx = red[0];
    __syncthreads();
    float sum = 0.0f;
#pragma unroll
    for (int i = 0; i < 4; ++i) { v[i] = __expf(v[i] - mx); sum += v[i]; }
    red[tid] = sum;
    __syncthreads();
    for (int st = 128; st > 0; st >>= 1) { if (tid < st) red[tid] += red[tid + st]; __syncthreads(); }
    float inv = 1.0f / red[0];
#pragma unroll
    for (int i = 0; i < 4; ++i) pc[(size_t)(tid + i * 256) * SEQ] = v[i] * inv;
}

// logvar head: out[r][j] = x[r] . Wlv[:,j] + blv[j]
__global__ void k_logvar(const float* __restrict__ X, const float* __restrict__ Wlv,
                         const float* __restrict__ blv, float* __restrict__ out) {
    int idx = blockIdx.x * blockDim.x + threadIdx.x;   // ROWS*2
    if (idx >= ROWS * 2) return;
    int j = idx & 1, r = idx >> 1;
    float acc = blv[j];
    const float* xr = X + (size_t)r * DIM;
    for (int d = 0; d < DIM; ++d) acc += xr[d] * Wlv[d * 2 + j];
    out[(size_t)r * 2 + j] = acc;
}

// ---------------------------------------------------------------------------
// Host orchestration
// ---------------------------------------------------------------------------
static inline size_t align256(size_t x) { return (x + 255) & ~(size_t)255; }

static void launch_gemm(const float* A, const _Float16* WT, const float* bias,
                        float* C, const float* resid, int M, int N, int K, int ldc,
                        float scale, hipStream_t s) {
    dim3 grid(N / 64, M / 128);
    k_gemm<<<grid, 256, 0, s>>>(A, WT, bias, C, resid, M, N, K, ldc, scale);
}

static void launch_convT(const float* W, _Float16* WT, int K, int N, hipStream_t s) {
    int count = K * N;
    k_convT<<<(count + 255) / 256, 256, 0, s>>>(W, WT, K, N);
}

extern "C" void kernel_launch(void* const* d_in, const int* in_sizes, int n_in,
                              void* d_out, int out_size, void* d_ws, size_t ws_size,
                              hipStream_t stream) {
    // ---- inputs (all f32) -------------------------------------------------
    const float* kpts0 = (const float*)d_in[0];
    const float* kpts1 = (const float*)d_in[1];
    const float* desc0 = (const float*)d_in[2];
    const float* desc1 = (const float*)d_in[3];
    const float* Wr    = (const float*)d_in[4];
    const float* Win   = (const float*)d_in[5];
    const float* b_in  = (const float*)d_in[6];
    const float* sWqkv = (const float*)d_in[7];
    const float* sbqkv = (const float*)d_in[8];
    const float* sWo   = (const float*)d_in[9];
    const float* sbo   = (const float*)d_in[10];
    const float* sW1   = (const float*)d_in[11];
    const float* sb1   = (const float*)d_in[12];
    const float* sg    = (const float*)d_in[13];
    const float* sbt   = (const float*)d_in[14];
    const float* sW2   = (const float*)d_in[15];
    const float* sb2   = (const float*)d_in[16];
    const float* cWqk  = (const float*)d_in[17];
    const float* cbqk  = (const float*)d_in[18];
    const float* cWv   = (const float*)d_in[19];
    const float* cbv   = (const float*)d_in[20];
    const float* cWo   = (const float*)d_in[21];
    const float* cbo   = (const float*)d_in[22];
    const float* cW1   = (const float*)d_in[23];
    const float* cb1   = (const float*)d_in[24];
    const float* cg    = (const float*)d_in[25];
    const float* cbt   = (const float*)d_in[26];
    const float* cW2   = (const float*)d_in[27];
    const float* cb2   = (const float*)d_in[28];
    const float* Wf    = (const float*)d_in[29];
    const float* bf    = (const float*)d_in[30];
    const float* Wlv   = (const float*)d_in[31];
    const float* blv   = (const float*)d_in[32];

    // ---- workspace carve --------------------------------------------------
    char* ws = (char*)d_ws;
    size_t off = 0;
    auto carve = [&](size_t bytes) -> void* { void* p = ws + off; off = align256(off + bytes); return p; };

    _Float16* whWin = (_Float16*)carve(65536u * 2);
    _Float16* whWf  = (_Float16*)carve(65536u * 2);
    _Float16* whQkv = (_Float16*)carve((size_t)LAYERS * 196608 * 2);
    _Float16* whSo  = (_Float16*)carve((size_t)LAYERS * 65536 * 2);
    _Float16* whS1  = (_Float16*)carve((size_t)LAYERS * 262144 * 2);
    _Float16* whS2  = (_Float16*)carve((size_t)LAYERS * 131072 * 2);
    _Float16* whQk  = (_Float16*)carve((size_t)LAYERS * 65536 * 2);
    _Float16* whV   = (_Float16*)carve((size_t)LAYERS * 65536 * 2);
    _Float16* whCo  = (_Float16*)carve((size_t)LAYERS * 65536 * 2);
    _Float16* whC1  = (_Float16*)carve((size_t)LAYERS * 262144 * 2);
    _Float16* whC2  = (_Float16*)carve((size_t)LAYERS * 131072 * 2);

    float* stats0 = (float*)carve(BATCH * 3 * 4);
    float* stats1 = (float*)carve(BATCH * 3 * 4);
    float* cos0 = (float*)carve((size_t)BATCH * SEQ * 64 * 4);
    float* sin0 = (float*)carve((size_t)BATCH * SEQ * 64 * 4);
    float* cos1 = (float*)carve((size_t)BATCH * SEQ * 64 * 4);
    float* sin1 = (float*)carve((size_t)BATCH * SEQ * 64 * 4);

    float* x0   = (float*)carve((size_t)ROWS * DIM * 4);
    float* x1   = (float*)carve((size_t)ROWS * DIM * 4);
    float* qkvb = (float*)carve((size_t)ROWS * 768 * 4);
    _Float16* hb0 = (_Float16*)carve((size_t)BATCH * HEADS * SEQ * 64 * 2);
    _Float16* hb1 = (_Float16*)carve((size_t)BATCH * HEADS * SEQ * 64 * 2);
    _Float16* hb2 = (_Float16*)carve((size_t)BATCH * HEADS * SEQ * 64 * 2);
    _Float16* hb3 = (_Float16*)carve((size_t)BATCH * HEADS * SEQ * 64 * 2);
    float* ctx  = (float*)carve((size_t)ROWS * DIM * 4);
    float* cat  = (float*)carve((size_t)ROWS * 512 * 4);
    float* h1   = (float*)carve((size_t)ROWS * 512 * 4);
    float* md0  = (float*)carve((size_t)ROWS * DIM * 4);
    float* md1  = (float*)carve((size_t)ROWS * DIM * 4);
    _Float16* md1h = (_Float16*)carve((size_t)ROWS * DIM * 2);
    float* sim  = (float*)carve((size_t)BATCH * SEQ * SEQ * 4);
    (void)ws_size; (void)in_sizes; (void)n_in; (void)out_size;

    // ---- weight conversion (f32 -> f16 transposed) ------------------------
    launch_convT(Win, whWin, DIM, DIM, stream);
    launch_convT(Wf,  whWf,  DIM, DIM, stream);
    for (int l = 0; l < LAYERS; ++l) {
        launch_convT(sWqkv + (size_t)l * 196608, whQkv + (size_t)l * 196608, DIM, 768, stream);
        launch_convT(sWo   + (size_t)l * 65536,  whSo  + (size_t)l * 65536,  DIM, DIM, stream);
        launch_convT(sW1   + (size_t)l * 262144, whS1  + (size_t)l * 262144, 512, 512, stream);
        launch_convT(sW2   + (size_t)l * 131072, whS2  + (size_t)l * 131072, 512, DIM, stream);
        launch_convT(cWqk  + (size_t)l * 65536,  whQk  + (size_t)l * 65536,  DIM, DIM, stream);
        launch_convT(cWv   + (size_t)l * 65536,  whV   + (size_t)l * 65536,  DIM, DIM, stream);
        launch_convT(cWo   + (size_t)l * 65536,  whCo  + (size_t)l * 65536,  DIM, DIM, stream);
        launch_convT(cW1   + (size_t)l * 262144, whC1  + (size_t)l * 262144, 512, 512, stream);
        launch_convT(cW2   + (size_t)l * 131072, whC2  + (size_t)l * 131072, 512, DIM, stream);
    }

    // ---- positional encodings --------------------------------------------
    k_stats<<<BATCH, 256, 0, stream>>>(kpts0, stats0);
    k_stats<<<BATCH, 256, 0, stream>>>(kpts1, stats1);
    k_posenc<<<BATCH * SEQ, 64, 0, stream>>>(kpts0, stats0, Wr, cos0, sin0);
    k_posenc<<<BATCH * SEQ, 64, 0, stream>>>(kpts1, stats1, Wr, cos1, sin1);

    // ---- input projection -------------------------------------------------
    launch_gemm(desc0, whWin, b_in, x0, nullptr, ROWS, DIM, DIM, DIM, 1.0f, stream);
    launch_gemm(desc1, whWin, b_in, x1, nullptr, ROWS, DIM, DIM, DIM, 1.0f, stream);

    // ---- transformer layers ----------------------------------------------
    auto run_self = [&](float* x, const float* cb, const float* sb, int l) {
        launch_gemm(x, whQkv + (size_t)l * 196608, sbqkv + (size_t)l * 768,
                    qkvb, nullptr, ROWS, 768, DIM, 768, 1.0f, stream);
        k_qkv_rope_split<<<(BATCH * HEADS * SEQ * 32) / 256, 256, 0, stream>>>(qkvb, cb, sb, hb0, hb1, hb2);
        k_attn<<<(BATCH * HEADS * (SEQ / 16)) / 8, 256, 0, stream>>>(hb0, hb1, hb2, ctx, 0.125f);
        k_copy_half<<<(ROWS * DIM) / 256, 256, 0, stream>>>(x, cat);
        launch_gemm(ctx, whSo + (size_t)l * 65536, sbo + (size_t)l * 256,
                    cat + 256, nullptr, ROWS, DIM, DIM, 512, 1.0f, stream);
        launch_gemm(cat, whS1 + (size_t)l * 262144, sb1 + (size_t)l * 512,
                    h1, nullptr, ROWS, 512, 512, 512, 1.0f, stream);
        k_ln_gelu<<<ROWS, 256, 0, stream>>>(h1, sg + (size_t)l * 512, sbt + (size_t)l * 512);
        launch_gemm(h1, whS2 + (size_t)l * 131072, sb2 + (size_t)l * 256,
                    x, x, ROWS, DIM, 512, DIM, 1.0f, stream);
    };

    auto run_cross_branch = [&](float* x, int l) {   // msg is in ctx
        k_copy_half<<<(ROWS * DIM) / 256, 256, 0, stream>>>(x, cat);
        launch_gemm(ctx, whCo + (size_t)l * 65536, cbo + (size_t)l * 256,
                    cat + 256, nullptr, ROWS, DIM, DIM, 512, 1.0f, stream);
        launch_gemm(cat, whC1 + (size_t)l * 262144, cb1 + (size_t)l * 512,
                    h1, nullptr, ROWS, 512, 512, 512, 1.0f, stream);
        k_ln_gelu<<<ROWS, 256, 0, stream>>>(h1, cg + (size_t)l * 512, cbt + (size_t)l * 512);
        launch_gemm(h1, whC2 + (size_t)l * 131072, cb2 + (size_t)l * 256,
                    x, x, ROWS, DIM, 512, DIM, 1.0f, stream);
    };

    for (int l = 0; l < LAYERS; ++l) {
        run_self(x0, cos0, sin0, l);
        run_self(x1, cos1, sin1, l);

        // cross block: qk0, qk1, v0^T, v1^T as f16 head buffers
        const int SPLIT_GRID = (BATCH * HEADS * SEQ * 64) / 256;
        launch_gemm(x0, whQk + (size_t)l * 65536, cbqk + (size_t)l * 256, ctx, nullptr, ROWS, DIM, DIM, DIM, 1.0f, stream);
        k_heads_split<false><<<SPLIT_GRID, 256, 0, stream>>>(ctx, hb0);
        launch_gemm(x1, whQk + (size_t)l * 65536, cbqk + (size_t)l * 256, ctx, nullptr, ROWS, DIM, DIM, DIM, 1.0f, stream);
        k_heads_split<false><<<SPLIT_GRID, 256, 0, stream>>>(ctx, hb1);
        launch_gemm(x0, whV + (size_t)l * 65536, cbv + (size_t)l * 256, ctx, nullptr, ROWS, DIM, DIM, DIM, 1.0f, stream);
        k_heads_split<true><<<SPLIT_GRID, 256, 0, stream>>>(ctx, hb2);
        launch_gemm(x1, whV + (size_t)l * 65536, cbv + (size_t)l * 256, ctx, nullptr, ROWS, DIM, DIM, DIM, 1.0f, stream);
        k_heads_split<true><<<SPLIT_GRID, 256, 0, stream>>>(ctx, hb3);

        const int ATTN_GRID = (BATCH * HEADS * (SEQ / 16)) / 8;
        // m0 = softmax_rows(qk0 qk1^T) v1 ; then x0 branch
        k_attn<<<ATTN_GRID, 256, 0, stream>>>(hb0, hb1, hb3, ctx, 0.125f);
        run_cross_branch(x0, l);
        // m1 = softmax_cols => attention with roles swapped: Q=qk1, K=qk0, V=v0
        k_attn<<<ATTN_GRID, 256, 0, stream>>>(hb1, hb0, hb2, ctx, 0.125f);
        run_cross_branch(x1, l);
    }

    // ---- final matching head ---------------------------------------------
    launch_gemm(x0, whWf, bf, md0, nullptr, ROWS, DIM, DIM, DIM, 0.25f, stream);  // / 256^0.25
    launch_gemm(x1, whWf, bf, md1, nullptr, ROWS, DIM, DIM, DIM, 0.25f, stream);
    k_conv16<<<(ROWS * DIM) / 256, 256, 0, stream>>>(md1, md1h, ROWS * DIM);
    for (int b = 0; b < BATCH; ++b) {
        launch_gemm(md0 + (size_t)b * SEQ * DIM, md1h + (size_t)b * SEQ * DIM, nullptr,
                    sim + (size_t)b * SEQ * SEQ, nullptr, SEQ, SEQ, DIM, SEQ, 1.0f, stream);
    }

    float* out = (float*)d_out;
    const size_t P_ELEMS = (size_t)BATCH * SEQ * SEQ;   // 2097152
    k_softmax_rows<<<BATCH * SEQ, 256, 0, stream>>>(sim, out);
    k_softmax_cols<<<BATCH * SEQ, 256, 0, stream>>>(sim, out + P_ELEMS);
    k_logvar<<<(ROWS * 2 + 255) / 256, 256, 0, stream>>>(x0, Wlv, blv, out + 2 * P_ELEMS);
    k_logvar<<<(ROWS * 2 + 255) / 256, 256, 0, stream>>>(x1, Wlv, blv, out + 2 * P_ELEMS + (size_t)ROWS * 2);
}
